// HL_HGCNN_pepfunc_dense_int3_attpool_87247965651035
// MI455X (gfx1250) — matrix operand order repaired and checked by reading, not compiled, not run on hardware
//
#include <hip/hip_runtime.h>
#include <hip/hip_bf16.h>

// ---------------- static problem sizes ----------------
#define B_    128
#define N0_   256
#define E0_   512
#define N1_   64
#define E1_   128
#define NT0_  (B_*N0_)      // 32768
#define NS0_  (B_*E0_)      // 65536
#define NT1_  (B_*N1_)      // 8192
#define NS1_  (B_*E1_)      // 16384
#define NNZT0 (3*NT0_)
#define NNZS0 (3*NS0_)
#define NNZT1 (3*NT1_)
#define NNZS1 (3*NS1_)
#define KT_IN 29
#define KS_IN 23
#define BN_EPS 1e-5f

#define FACC  1
#define FRELU 2
#define FSIG  4

typedef __attribute__((ext_vector_type(16))) __bf16 v16bf;
typedef __attribute__((ext_vector_type(8)))  float  v8f;

union ABfrag { v16bf v; uint4 q[2]; unsigned short u[16]; };

#if defined(__has_builtin)
#if __has_builtin(__builtin_amdgcn_cvt_pk_bf16_f32)
#define HAVE_CVT_PK_BF16 1
#endif
#if __has_builtin(__builtin_amdgcn_perm)
#define HAVE_PERM 1
#endif
#endif

static __device__ __forceinline__ unsigned int pack2(float lo, float hi) {
#if defined(HAVE_CVT_PK_BF16)
  auto r = __builtin_amdgcn_cvt_pk_bf16_f32(lo, hi);     // v_cvt_pk_bf16_f32
  union { decltype(r) v; unsigned int u; } c; c.v = r;
  return c.u;
#elif defined(HAVE_PERM)
  // round-half-up bf16: add 0x8000, keep high halves; one v_perm_b32 packs both.
  union { float f; unsigned int u; } a, b;
  a.f = lo; b.f = hi;
  return __builtin_amdgcn_perm(b.u + 0x8000u, a.u + 0x8000u, 0x07060302u);
#else
  union { float f; unsigned int u; } a, b;
  a.f = lo; b.f = hi;
  unsigned int l = (a.u + 0x8000u) >> 16;
  unsigned int h = (b.u + 0x8000u) & 0xffff0000u;
  return h | l;
#endif
}

// ======== fast WMMA GEMM: M%32==0, N%64==0, K%32==0, lda%4==0, A 16B-aligned ========
// tile 32(M) x 64(N), BK=32; register double-buffered global->LDS pipeline.
__global__ __launch_bounds__(128)
void k_gemm_fast(const float* __restrict__ A, int lda,
                 const float* __restrict__ W, int ldw,
                 const float* __restrict__ bias,
                 float* __restrict__ C, int ldc,
                 int M, int N, int K, int flags)
{
  __shared__ unsigned short As[32][40];    // [m][k]  80B rows, 16B aligned
  __shared__ unsigned short BsT[64][40];   // [n][k]  transposed W tile

  const int tid  = threadIdx.x;
  const int m0   = blockIdx.y << 5;
  const int n0   = blockIdx.x << 6;
  const int lane = tid & 31;
  const int wave = tid >> 5;
  const int wm   = (wave >> 1) << 4;
  const int wn   = (wave & 1)  << 4;
  const int lr   = lane & 15;
  const int kh   = lane >> 4;

  v8f acc0, acc1;
#pragma unroll
  for (int i = 0; i < 8; ++i) { acc0[i] = 0.0f; acc1[i] = 0.0f; }

  const int ar = tid >> 2;
  const int ac = (tid & 3) << 3;
  const int bn = tid & 63;
  const int bh = tid >> 6;
  const int kt = K >> 5;

  const float* arow = A + (size_t)(m0 + ar) * lda + ac;
  const float* wp   = W + (size_t)(bh << 4) * ldw + n0 + bn;
  const size_t wstep = (size_t)ldw;

  // ---- prologue: load tile 0 into registers ----
  float4 ra0, ra1; float rw[16];
  {
    const float4* ap = (const float4*)arow;
    ra0 = ap[0]; ra1 = ap[1];
    const float* p = wp;
#pragma unroll
    for (int j = 0; j < 16; ++j) { rw[j] = *p; p += wstep; }
    arow += 32; wp += wstep << 5;
  }

  for (int t = 0; t < kt; ++t) {
    // ---- pack current register tile -> LDS ----
    uint4 pa; pa.x = pack2(ra0.x, ra0.y); pa.y = pack2(ra0.z, ra0.w);
              pa.z = pack2(ra1.x, ra1.y); pa.w = pack2(ra1.z, ra1.w);
    *(uint4*)&As[ar][ac] = pa;
    uint4 p0, p1;
    p0.x = pack2(rw[0], rw[1]);   p0.y = pack2(rw[2], rw[3]);
    p0.z = pack2(rw[4], rw[5]);   p0.w = pack2(rw[6], rw[7]);
    p1.x = pack2(rw[8], rw[9]);   p1.y = pack2(rw[10], rw[11]);
    p1.z = pack2(rw[12], rw[13]); p1.w = pack2(rw[14], rw[15]);
    *(uint4*)&BsT[bn][(bh << 4) + 0] = p0;
    *(uint4*)&BsT[bn][(bh << 4) + 8] = p1;
    __syncthreads();

    // ---- issue next tile's global loads (overlap with WMMA below) ----
    if (t + 1 < kt) {                       // uniform branch: EXEC untouched
      const float4* ap = (const float4*)arow;
      ra0 = ap[0]; ra1 = ap[1];
      const float* p = wp;
#pragma unroll
      for (int j = 0; j < 16; ++j) { rw[j] = *p; p += wstep; }
      arow += 32; wp += wstep << 5;
    }

    // ---- fragments + WMMA ----
    ABfrag a, b0, b1;
    a.q[0]  = *(const uint4*)&As[wm + lr][(kh << 4)];
    a.q[1]  = *(const uint4*)&As[wm + lr][(kh << 4) + 8];
    b0.q[0] = *(const uint4*)&BsT[wn + lr][(kh << 4)];
    b0.q[1] = *(const uint4*)&BsT[wn + lr][(kh << 4) + 8];
    b1.q[0] = *(const uint4*)&BsT[wn + 32 + lr][(kh << 4)];
    b1.q[1] = *(const uint4*)&BsT[wn + 32 + lr][(kh << 4) + 8];
    acc0 = __builtin_amdgcn_wmma_f32_16x16x32_bf16(
        false, a.v, false, b0.v, (short)0, acc0, false, false);
    acc1 = __builtin_amdgcn_wmma_f32_16x16x32_bf16(
        false, a.v, false, b1.v, (short)0, acc1, false, false);
    __syncthreads();
  }

  // ---- epilogue ----
  const int c0 = n0 + wn + lr;
  const int c1 = c0 + 32;
  float bias0 = 0.0f, bias1 = 0.0f;
  if (bias) { bias0 = bias[c0]; bias1 = bias[c1]; }
#pragma unroll
  for (int r = 0; r < 8; ++r) {
    int row = m0 + wm + (kh << 3) + r;
    float v0 = acc0[r] + bias0, v1 = acc1[r] + bias1;
    if (flags & FACC) {
      v0 += C[(size_t)row * ldc + c0];
      v1 += C[(size_t)row * ldc + c1];
    }
    if (flags & FRELU) { v0 = fmaxf(v0, 0.0f); v1 = fmaxf(v1, 0.0f); }
    if (flags & FSIG)  { v0 = 1.0f / (1.0f + __expf(-v0)); v1 = 1.0f / (1.0f + __expf(-v1)); }
    C[(size_t)row * ldc + c0] = v0;
    C[(size_t)row * ldc + c1] = v1;
  }
}

// ======== guarded WMMA GEMM (value-masked, branchless): M%32==0, any N/K/lda ========
__global__ __launch_bounds__(128)
void k_gemm_any(const float* __restrict__ A, int lda,
                const float* __restrict__ W, int ldw,
                const float* __restrict__ bias,
                float* __restrict__ C, int ldc,
                int M, int N, int K, int flags)
{
  __shared__ unsigned short As[32][40];
  __shared__ unsigned short BsT[32][40];

  const int tid  = threadIdx.x;
  const int m0   = blockIdx.y << 5;
  const int n0   = blockIdx.x << 5;
  const int lane = tid & 31;
  const int wave = tid >> 5;
  const int wm   = (wave >> 1) << 4;
  const int wn   = (wave & 1)  << 4;
  const int lr   = lane & 15;
  const int kh   = lane >> 4;

  v8f acc;
#pragma unroll
  for (int i = 0; i < 8; ++i) acc[i] = 0.0f;

  const int ar = tid >> 2;
  const int ac = (tid & 3) << 3;
  const int bn = tid & 31;
  const int bq = tid >> 5;
  const int kt = (K + 31) >> 5;

  const int nn = n0 + bn;
  const int nc = nn < N ? nn : N - 1;
  const float* arow = A + (size_t)(m0 + ar) * lda + ac;
  const float* wp   = W + (size_t)(bq << 3) * ldw + nc;
  const size_t wstep = (size_t)ldw;

  for (int t = 0; t < kt; ++t) {
    const int k0 = t << 5;
    float av[8];
#pragma unroll
    for (int c2 = 0; c2 < 8; ++c2) {
      int k = k0 + ac + c2;
      float val = arow[c2];
      av[c2] = (k < K) ? val : 0.0f;
    }
    uint4 pa;
    pa.x = pack2(av[0], av[1]); pa.y = pack2(av[2], av[3]);
    pa.z = pack2(av[4], av[5]); pa.w = pack2(av[6], av[7]);
    *(uint4*)&As[ar][ac] = pa;

    float wv[8];
    {
      const float* p = wp;
#pragma unroll
      for (int j = 0; j < 8; ++j) {
        int k = k0 + (bq << 3) + j;
        float val = *p; p += wstep;
        wv[j] = (k < K && nn < N) ? val : 0.0f;
      }
    }
    uint4 pb;
    pb.x = pack2(wv[0], wv[1]); pb.y = pack2(wv[2], wv[3]);
    pb.z = pack2(wv[4], wv[5]); pb.w = pack2(wv[6], wv[7]);
    *(uint4*)&BsT[bn][bq << 3] = pb;
    arow += 32;
    wp   += wstep << 5;
    __syncthreads();

    ABfrag a, b;
    a.q[0] = *(const uint4*)&As[wm + lr][(kh << 4)];
    a.q[1] = *(const uint4*)&As[wm + lr][(kh << 4) + 8];
    b.q[0] = *(const uint4*)&BsT[wn + lr][(kh << 4)];
    b.q[1] = *(const uint4*)&BsT[wn + lr][(kh << 4) + 8];
    acc = __builtin_amdgcn_wmma_f32_16x16x32_bf16(
        false, a.v, false, b.v, (short)0, acc, false, false);
    __syncthreads();
  }

#pragma unroll
  for (int r = 0; r < 8; ++r) {
    int row = m0 + wm + (kh << 3) + r;
    int col = n0 + wn + lr;
    if (col < N) {
      float v = acc[r];
      if (bias) v += bias[col];
      if (flags & FACC)  v += C[(size_t)row * ldc + col];
      if (flags & FRELU) v = fmaxf(v, 0.0f);
      if (flags & FSIG)  v = 1.0f / (1.0f + __expf(-v));
      C[(size_t)row * ldc + col] = v;
    }
  }
}

// ---------------- 2D row x col helpers (no div/mod; row index block-uniform) ----------------
#define CBLK 64

__global__ void k_fill(float* __restrict__ p, float val, long long n) {
  for (long long i = blockIdx.x * (long long)blockDim.x + threadIdx.x; i < n;
       i += (long long)gridDim.x * blockDim.x) p[i] = val;
}

__global__ void k_deg(const int* __restrict__ u, const int* __restrict__ v,
                      int ne, float* __restrict__ D) {
  for (long long i = blockIdx.x * (long long)blockDim.x + threadIdx.x; i < ne;
       i += (long long)gridDim.x * blockDim.x) {
    atomicAdd(&D[u[i]], 1.0f);
    atomicAdd(&D[v[i]], 1.0f);
  }
}

__global__ void k_s2t(const float* __restrict__ XS, int lds,
                      const int* __restrict__ u, const int* __restrict__ v,
                      int cols, float* __restrict__ OUT) {
  int e = blockIdx.x;
  int c = blockIdx.y * CBLK + threadIdx.x;
  if (c >= cols) return;
  float val = XS[(size_t)e * lds + c];
  atomicAdd(&OUT[(size_t)u[e] * cols + c], val);
  atomicAdd(&OUT[(size_t)v[e] * cols + c], val);
}

__global__ void k_divD(float* __restrict__ X, const float* __restrict__ D, int cols) {
  int r = blockIdx.x;
  int c = blockIdx.y * CBLK + threadIdx.x;
  if (c >= cols) return;
  X[(size_t)r * cols + c] /= D[r];
}

__global__ void k_t2s(const float* __restrict__ XT, int ldt,
                      const int* __restrict__ u, const int* __restrict__ v,
                      int cols, float* __restrict__ OUT) {
  int e = blockIdx.x;
  int c = blockIdx.y * CBLK + threadIdx.x;
  if (c >= cols) return;
  OUT[(size_t)e * cols + c] =
      0.5f * (XT[(size_t)u[e] * ldt + c] + XT[(size_t)v[e] * ldt + c]);
}

__global__ void k_copy_slice(const float* __restrict__ S, int lds,
                             float* __restrict__ D, int ldd, int cols) {
  int r = blockIdx.x;
  int c = blockIdx.y * CBLK + threadIdx.x;
  if (c >= cols) return;
  D[(size_t)r * ldd + c] = S[(size_t)r * lds + c];
}

__global__ void k_lag(const float* __restrict__ X, int ldx,
                      const int* __restrict__ src, const int* __restrict__ dst,
                      const float* __restrict__ w, int cols, float* __restrict__ Y) {
  int z = blockIdx.x;
  int c = blockIdx.y * CBLK + threadIdx.x;
  if (c >= cols) return;
  atomicAdd(&Y[(size_t)dst[z] * cols + c], -w[z] * X[(size_t)src[z] * ldx + c]);
}

__global__ __launch_bounds__(256)
void k_colstats(const float* __restrict__ X, int ld, int rows,
                float* __restrict__ mean, float* __restrict__ var) {
  __shared__ float ssum[256], ssq[256];
  int c = blockIdx.x;
  float s = 0.0f, q = 0.0f;
  for (int r = threadIdx.x; r < rows; r += 256) {
    float v = X[(size_t)r * ld + c];
    s += v; q += v * v;
  }
  ssum[threadIdx.x] = s; ssq[threadIdx.x] = q;
  __syncthreads();
  for (int st = 128; st > 0; st >>= 1) {
    if (threadIdx.x < st) {
      ssum[threadIdx.x] += ssum[threadIdx.x + st];
      ssq[threadIdx.x]  += ssq[threadIdx.x + st];
    }
    __syncthreads();
  }
  if (threadIdx.x == 0) {
    float m = ssum[0] / rows;
    mean[c] = m;
    var[c]  = ssq[0] / rows - m * m;
  }
}

__global__ void k_bnrelu(float* __restrict__ X, int ld, int cols,
                         const float* __restrict__ mean, const float* __restrict__ var,
                         const float* __restrict__ g, const float* __restrict__ be) {
  int r = blockIdx.x;
  int c = blockIdx.y * CBLK + threadIdx.x;
  if (c >= cols) return;
  float v = X[(size_t)r * ld + c];
  v = (v - mean[c]) * rsqrtf(var[c] + BN_EPS) * g[c] + be[c];
  X[(size_t)r * ld + c] = fmaxf(v, 0.0f);
}

__global__ void k_pool(const float* __restrict__ X, int ldx,
                       const float* __restrict__ scale, const int* __restrict__ pos,
                       int cols, float* __restrict__ OUT, int ldo) {
  int r = blockIdx.x;
  int c = blockIdx.y * CBLK + threadIdx.x;
  if (c >= cols) return;
  atomicAdd(&OUT[(size_t)pos[r] * ldo + c], X[(size_t)r * ldx + c] * scale[r]);
}

__global__ void k_cnt(const int* __restrict__ pos, int rows, float* __restrict__ cnt) {
  for (long long i = blockIdx.x * (long long)blockDim.x + threadIdx.x; i < rows;
       i += (long long)gridDim.x * blockDim.x)
    atomicAdd(&cnt[pos[i]], 1.0f);
}

__global__ void k_divcnt(float* __restrict__ X, int ld, int cols,
                         const float* __restrict__ cnt) {
  int r = blockIdx.x;
  int c = blockIdx.y * CBLK + threadIdx.x;
  if (c >= cols) return;
  X[(size_t)r * ld + c] /= fmaxf(cnt[r], 1.0f);
}

__global__ void k_segmean(const float* __restrict__ X, int ldx, int rps,
                          int cols, float* __restrict__ OUT, int ldo, int coff) {
  int s = blockIdx.x;
  int c = blockIdx.y * CBLK + threadIdx.x;
  if (c >= cols) return;
  float acc = 0.0f;
  for (int r = 0; r < rps; ++r) acc += X[((size_t)s * rps + r) * ldx + c];
  OUT[(size_t)s * ldo + coff + c] = acc / (float)rps;
}

__global__ void k_head(const float* __restrict__ G, const float* __restrict__ W,
                       const float* __restrict__ bias, float* __restrict__ OUT) {
  int i = blockIdx.x * blockDim.x + threadIdx.x;
  if (i >= B_ * 10) return;
  int b = i / 10, cls = i % 10;
  float acc = bias[cls];
  const float* p = W + cls;
  for (int k = 0; k < 1024; ++k) { acc += G[(size_t)b * 1024 + k] * (*p); p += 10; }
  OUT[i] = acc;
}

// ---------------- host orchestration ----------------
struct LinP { const float* W; const float* b; };
struct BnP  { const float* be; const float* g; };
struct BlockP { BnP cs_bn; LinP cs0, cs1; BnP ct_bn; LinP ct0, ct1; LinP is, it; };

static inline dim3 ewGrid(long long n) {
  long long b = (n + 255) / 256;
  if (b > 1048576) b = 1048576;
  if (b < 1) b = 1;
  return dim3((unsigned)b);
}
static inline dim3 g2(int rows, int cols) {
  return dim3((unsigned)rows, (unsigned)((cols + CBLK - 1) / CBLK));
}

extern "C" void kernel_launch(void* const* d_in, const int* in_sizes, int n_in,
                              void* d_out, int out_size, void* d_ws, size_t ws_size,
                              hipStream_t stream) {
  (void)in_sizes; (void)n_in; (void)out_size; (void)ws_size;

  const float* x_t   = (const float*)d_in[0];
  const float* x_s   = (const float*)d_in[1];
  const int*   pos_t = (const int*)d_in[2];
  const int*   pos_s = (const int*)d_in[3];
  const int*   ei0   = (const int*)d_in[4];
  const int*   eit0  = (const int*)d_in[5];
  const float* ewt0  = (const float*)d_in[6];
  const int*   eis0  = (const int*)d_in[7];
  const float* ews0  = (const float*)d_in[8];
  const int*   ei1   = (const int*)d_in[9];
  const int*   eit1  = (const int*)d_in[10];
  const float* ewt1  = (const float*)d_in[11];
  const int*   eis1  = (const int*)d_in[12];
  const float* ews1  = (const float*)d_in[13];
  const float* P     = (const float*)d_in[14];

  static const int F8[8]   = {64, 64, 128, 128, 256, 256, 512, 512};
  static const int DIN8[8] = {64, 128, 192, 320, 448, 704, 960, 1472};
  size_t off = 0;
  auto take = [&](size_t n) { const float* r = P + off; off += n; return r; };

  LinP att_s, att_t;
  att_s.W = take(384); att_s.b = take(1);
  att_t.W = take(384); att_t.b = take(1);
  BlockP blk[8];
  for (int i = 0; i < 8; ++i) {
    size_t f = F8[i], din = DIN8[i];
    blk[i].cs_bn.be = take(f); blk[i].cs_bn.g = take(f);
    blk[i].cs0.W = take(f * f); blk[i].cs0.b = take(f);
    blk[i].cs1.W = take(f * f); blk[i].cs1.b = take(f);
    blk[i].ct_bn.be = take(f); blk[i].ct_bn.g = take(f);
    blk[i].ct0.W = take(f * f); blk[i].ct0.b = take(f);
    blk[i].ct1.W = take(f * f); blk[i].ct1.b = take(f);
    blk[i].is.W = take(2 * din * f); blk[i].is.b = take(f);
    blk[i].it.W = take(2 * din * f); blk[i].it.b = take(f);
  }
  BnP init_s_bn; LinP init_s_l;
  init_s_bn.be = take(64); init_s_bn.g = take(64);
  init_s_l.W = take((size_t)KS_IN * 64); init_s_l.b = take(64);
  BnP init_t_bn; LinP init_t_l;
  init_t_bn.be = take(64); init_t_bn.g = take(64);
  init_t_l.W = take((size_t)KT_IN * 64); init_t_l.b = take(64);
  LinP outL; outL.W = take(1024 * 10); outL.b = take(10);

  float* w = (float*)d_ws; size_t wo = 0;
  auto alloc = [&](size_t n) { float* p = w + wo; wo += n; return p; };
  float* XT0  = alloc((size_t)NT0_ * 192);
  float* XS0  = alloc((size_t)NS0_ * 192);
  float* XT1  = alloc((size_t)NT1_ * 1984);
  float* XS1  = alloc((size_t)NS1_ * 1984);
  float* SCRT = alloc((size_t)NT1_ * 1472);
  float* SCRS = alloc((size_t)NS1_ * 1472);
  float* TMPT = alloc((size_t)NT1_ * 512);
  float* TMPS = alloc((size_t)NS1_ * 512);
  float* Dbuf = alloc(NT0_);
  float* ATb  = alloc(NT0_);
  float* ASb  = alloc(NS0_);
  float* CNT  = alloc(NS1_);
  float* STAT = alloc(1024);
  float* G    = alloc((size_t)B_ * 1024);

  auto gemmF = [&](const float* A, int lda, const float* W_, int ldw,
                   const float* bias, float* C, int ldc, int M, int N, int K, int flags) {
    dim3 grid(N >> 6, M >> 5);
    k_gemm_fast<<<grid, 128, 0, stream>>>(A, lda, W_, ldw, bias, C, ldc, M, N, K, flags);
  };
  auto gemmA = [&](const float* A, int lda, const float* W_, int ldw,
                   const float* bias, float* C, int ldc, int M, int N, int K, int flags) {
    dim3 grid((N + 31) / 32, M >> 5);
    k_gemm_any<<<grid, 128, 0, stream>>>(A, lda, W_, ldw, bias, C, ldc, M, N, K, flags);
  };
  auto fill = [&](float* p, float v, long long n) {
    k_fill<<<ewGrid(n), 256, 0, stream>>>(p, v, n);
  };
  auto bn = [&](float* X, int ld, int rows, int cols, const BnP& b) {
    k_colstats<<<cols, 256, 0, stream>>>(X, ld, rows, STAT, STAT + 512);
    k_bnrelu<<<g2(rows, cols), CBLK, 0, stream>>>(X, ld, cols, STAT, STAT + 512, b.g, b.be);
  };

  // ================= forward =================
  gemmA(x_t, KT_IN, init_t_l.W, 64, init_t_l.b, XT0, 192, NT0_, 64, KT_IN, 0);
  bn(XT0, 192, NT0_, 64, init_t_bn);
  gemmA(x_s, KS_IN, init_s_l.W, 64, init_s_l.b, XS0, 192, NS0_, 64, KS_IN, 0);
  bn(XS0, 192, NS0_, 64, init_s_bn);

  int rowsT = NT0_, rowsS = NS0_, ldT = 192, ldS = 192;
  float *XT = XT0, *XS = XS0;
  const int *u = ei0, *v = ei0 + NS0_;
  int ne = NS0_;
  const int *ts = eit0, *td = eit0 + NNZT0; const float* tw = ewt0; int tn = NNZT0;
  const int *ss = eis0, *sd = eis0 + NNZS0; const float* sw = ews0; int sn = NNZS0;
  fill(Dbuf, 1e-6f, rowsT);
  k_deg<<<ewGrid(ne), 256, 0, stream>>>(u, v, ne, Dbuf);

  int din = 64, bi = 0;
  for (int lvl = 0; lvl < 4; ++lvl) {
    int f = F8[2 * lvl];
    for (int ch = 0; ch < 2; ++ch) {
      const BlockP& bk = blk[bi++];
      fill(SCRT, 0.0f, (long long)rowsT * din);
      k_s2t<<<g2(ne, din), CBLK, 0, stream>>>(XS, ldS, u, v, din, SCRT);
      k_divD<<<g2(rowsT, din), CBLK, 0, stream>>>(SCRT, Dbuf, din);
      k_t2s<<<g2(ne, din), CBLK, 0, stream>>>(XT, ldT, u, v, din, SCRS);
      gemmF(XT, ldT, bk.it.W, f, bk.it.b, TMPT, f, rowsT, f, din, 0);
      gemmF(SCRT, din, bk.it.W + (size_t)din * f, f, nullptr, TMPT, f, rowsT, f, din, FACC | FRELU);
      gemmF(XS, ldS, bk.is.W, f, bk.is.b, TMPS, f, rowsS, f, din, 0);
      gemmF(SCRS, din, bk.is.W + (size_t)din * f, f, nullptr, TMPS, f, rowsS, f, din, FACC | FRELU);
      float* CT = XT + din;
      gemmF(TMPT, f, bk.ct0.W, f, bk.ct0.b, CT, ldT, rowsT, f, f, 0);
      k_copy_slice<<<g2(rowsT, f), CBLK, 0, stream>>>(TMPT, f, SCRT, f, f);
      k_lag<<<g2(tn, f), CBLK, 0, stream>>>(TMPT, f, ts, td, tw, f, SCRT);
      gemmF(SCRT, f, bk.ct1.W, f, bk.ct1.b, CT, ldT, rowsT, f, f, FACC);
      bn(CT, ldT, rowsT, f, bk.ct_bn);
      float* CS = XS + din;
      gemmF(TMPS, f, bk.cs0.W, f, bk.cs0.b, CS, ldS, rowsS, f, f, 0);
      k_copy_slice<<<g2(rowsS, f), CBLK, 0, stream>>>(TMPS, f, SCRS, f, f);
      k_lag<<<g2(sn, f), CBLK, 0, stream>>>(TMPS, f, ss, sd, sw, f, SCRS);
      gemmF(SCRS, f, bk.cs1.W, f, bk.cs1.b, CS, ldS, rowsS, f, f, FACC);
      bn(CS, ldS, rowsS, f, bk.cs_bn);
      din += f;
    }
    if (lvl == 0) {
      fill(SCRT, 0.0f, (long long)rowsT * din);
      k_s2t<<<g2(ne, din), CBLK, 0, stream>>>(XS, ldS, u, v, din, SCRT);
      k_divD<<<g2(rowsT, din), CBLK, 0, stream>>>(SCRT, Dbuf, din);
      k_t2s<<<g2(ne, din), CBLK, 0, stream>>>(XT, ldT, u, v, din, SCRS);
      gemmA(XT, ldT, att_t.W, 1, att_t.b, ATb, 1, rowsT, 1, din, 0);
      gemmA(SCRT, din, att_t.W + din, 1, nullptr, ATb, 1, rowsT, 1, din, FACC | FSIG);
      gemmA(XS, ldS, att_s.W, 1, att_s.b, ASb, 1, rowsS, 1, din, 0);
      gemmA(SCRS, din, att_s.W + din, 1, nullptr, ASb, 1, rowsS, 1, din, FACC | FSIG);
      fill(XT1, 0.0f, (long long)NT1_ * 1984);
      fill(CNT, 0.0f, NT1_);
      k_pool<<<g2(rowsT, din), CBLK, 0, stream>>>(XT, ldT, ATb, pos_t, din, XT1, 1984);
      k_cnt<<<ewGrid(rowsT), 256, 0, stream>>>(pos_t, rowsT, CNT);
      k_divcnt<<<g2(NT1_, din), CBLK, 0, stream>>>(XT1, 1984, din, CNT);
      fill(XS1, 0.0f, (long long)NS1_ * 1984);
      fill(CNT, 0.0f, NS1_);
      k_pool<<<g2(rowsS, din), CBLK, 0, stream>>>(XS, ldS, ASb, pos_s, din, XS1, 1984);
      k_cnt<<<ewGrid(rowsS), 256, 0, stream>>>(pos_s, rowsS, CNT);
      k_divcnt<<<g2(NS1_, din), CBLK, 0, stream>>>(XS1, 1984, din, CNT);
      rowsT = NT1_; rowsS = NS1_; XT = XT1; XS = XS1; ldT = 1984; ldS = 1984;
      u = ei1; v = ei1 + NS1_; ne = NS1_;
      ts = eit1; td = eit1 + NNZT1; tw = ewt1; tn = NNZT1;
      ss = eis1; sd = eis1 + NNZS1; sw = ews1; sn = NNZS1;
      fill(Dbuf, 1e-6f, rowsT);
      k_deg<<<ewGrid(ne), 256, 0, stream>>>(u, v, ne, Dbuf);
    }
  }

  k_segmean<<<g2(B_, 512), CBLK, 0, stream>>>(XS1 + 1472, 1984, E1_, 512, G, 1024, 0);
  k_segmean<<<g2(B_, 512), CBLK, 0, stream>>>(XT1 + 1472, 1984, N1_, 512, G, 1024, 512);
  k_head<<<(B_ * 10 + 127) / 128, 128, 0, stream>>>(G, outL.W, outL.b, (float*)d_out);
}
